// SNNAssociativeLanguageModel_55576876810897
// MI455X (gfx1250) — matrix operand order repaired and checked by reading, not compiled, not run on hardware
//
#include <hip/hip_runtime.h>

// ---------------------------------------------------------------------------
// SNN language model forward for MI455X (gfx1250, wave32, WMMA).
// GEMMs on v_wmma_f32_16x16x32_bf16, bf16 operands (weights pre-transposed).
// Tile staging:
//   - layer GEMMs: GLOBAL_LOAD_ASYNC_TO_LDS_B128 (ASYNCcnt) double-buffered
//   - logits GEMM: TENSOR_LOAD_TO_LDS (TDM, TENSORcnt) with HW LDS padding
// ---------------------------------------------------------------------------

typedef __attribute__((ext_vector_type(16))) __bf16   v16bf;
typedef __attribute__((ext_vector_type(8)))  __bf16   v8bf;
typedef __attribute__((ext_vector_type(8)))  float    v8f;
typedef __attribute__((ext_vector_type(4)))  unsigned u32x4;
typedef __attribute__((ext_vector_type(8)))  unsigned u32x8;

#define T_DIM 512
#define B_DIM 8
#define H_DIM 256
#define V_DIM 32000
#define L_DIM 3
#define M_DIM (T_DIM * B_DIM)      // 4096 rows in every GEMM

// GEMM tiling: block = 128x128, K-step 32, 8 waves (256 thr), wave = 64x32.
#define BM 128
#define BN 128
#define BK 32
#define LDSS 40                    // bf16 elems per LDS row (32 + 8 pad = 80B)
#define SEG  (BM * LDSS)           // one operand tile in bf16 elements (10240B)
#define KSTEPS (H_DIM / BK)        // 8

// ---------------------------------------------------------------------------
// CDNA5 async global->LDS copy (ASYNCcnt-tracked).
// ---------------------------------------------------------------------------
__device__ __forceinline__ void async_b128(unsigned lds_addr, unsigned goff,
                                           unsigned long long base) {
    asm volatile("global_load_async_to_lds_b128 %0, %1, %2"
                 :: "v"(lds_addr), "v"(goff), "s"(base) : "memory");
}
__device__ __forceinline__ void wait_async_le4() {
    asm volatile("s_wait_asynccnt 0x4" ::: "memory");
}
__device__ __forceinline__ void wait_async_0() {
    asm volatile("s_wait_asynccnt 0x0" ::: "memory");
}

// ---------------------------------------------------------------------------
// CDNA5 Tensor Data Mover: 2D tile (tile_dim0=32 elems x tile_dim1=128 rows,
// data_size=2B, tensor_dim0_stride=256) -> LDS with pad 16B after every 64B
// (reproduces the 80B conflict-free LDS row stride in hardware).
// ---------------------------------------------------------------------------
__device__ __forceinline__ void tdm_tile_2d(unsigned lds_addr,
                                            unsigned long long gaddr) {
    u32x4 g0;
    g0[0] = 1u;                                                 // count=1
    g0[1] = lds_addr;                                           // LDS byte addr
    g0[2] = (unsigned)gaddr;                                    // addr[31:0]
    g0[3] = ((unsigned)(gaddr >> 32) & 0x01FFFFFFu) | (2u << 30); // addr hi, type=2
    u32x8 g1;
    g1[0] = (1u << 16)          // data_size = 2 bytes
          | (1u << 20)          // pad_enable
          | (3u << 22)          // pad_interval: 16 DWORDs = 64B
          | (3u << 25);         // pad_amount: 4 DWORDs = 16B
    g1[1] = (256u & 0xFFFFu) << 16;       // tensor_dim0 = 256 (bits 79:48 lo)
    g1[2] = 0u;                           // dim0 hi | tensor_dim1 lo (dim1=1<<20)
    g1[3] = (1u << 4) | (32u << 16);      // tensor_dim1 hi = 16, tile_dim0 = 32
    g1[4] = 128u;                         // tile_dim1 = 128 rows (tile_dim2=0)
    g1[5] = 256u;                         // tensor_dim0_stride = 256 elems
    g1[6] = 0u;
    g1[7] = 0u;
    asm volatile("tensor_load_to_lds %0, %1" :: "s"(g0), "s"(g1) : "memory");
}

// ---------------------------------------------------------------------------
// Embedding gather -> bf16: h[m,hh] = temb[tokens[m],hh] + pemb[m/B,hh]
// ---------------------------------------------------------------------------
__global__ __launch_bounds__(256)
void snn_embed_kernel(const int* __restrict__ tokens,
                      const float* __restrict__ temb,
                      const float* __restrict__ pemb,
                      __bf16* __restrict__ h) {
    int e  = blockIdx.x * 256 + threadIdx.x;   // 0 .. M*H-1
    int m  = e >> 8;
    int hh = e & (H_DIM - 1);
    int t  = m >> 3;                           // m / B
    int tok = tokens[m];
    h[e] = (__bf16)(temb[(size_t)tok * H_DIM + hh] + pemb[(size_t)t * H_DIM + hh]);
}

// ---------------------------------------------------------------------------
// Weight convert + transpose: W[K=256][N] f32 -> WT[N][256] bf16.
// ---------------------------------------------------------------------------
__global__ __launch_bounds__(256)
void snn_convt_kernel(const float* __restrict__ W, __bf16* __restrict__ WT, int N) {
    __shared__ float tile[32][33];
    int bn = blockIdx.x * 32;
    int bk = blockIdx.y * 32;
    int tx = threadIdx.x & 31;
    int ty = threadIdx.x >> 5;                 // 0..7
#pragma unroll
    for (int i = 0; i < 4; ++i) {
        int k = ty + i * 8;
        tile[k][tx] = W[(size_t)(bk + k) * N + bn + tx];
    }
    __syncthreads();
#pragma unroll
    for (int i = 0; i < 4; ++i) {
        int n = ty + i * 8;
        WT[(size_t)(bn + n) * H_DIM + bk + tx] = (__bf16)tile[tx][n];
    }
}

// ---------------------------------------------------------------------------
// Leaky scan + spike: cur(f32) -> spk(bf16). One thread per (b,hh) channel.
// ---------------------------------------------------------------------------
__global__ __launch_bounds__(256)
void snn_scan_spike_kernel(const float* __restrict__ cur,
                           __bf16* __restrict__ spk,
                           const float* __restrict__ beta) {
    int c  = blockIdx.x * 256 + threadIdx.x;   // 0 .. B*H-1 (2048)
    int b  = c >> 8;
    int hh = c & (H_DIM - 1);
    float bt  = beta[hh];
    float mem = 0.0f;
    for (int t = 0; t < T_DIM; ++t) {
        size_t idx = ((size_t)(t * B_DIM + b) << 8) + hh;
        mem = fmaf(bt, mem, cur[idx]);
        spk[idx] = (mem - 1.0f) > 0.0f ? (__bf16)1.0f : (__bf16)0.0f;
    }
}

// ---------------------------------------------------------------------------
// C[M,N] = A[M,256] * BT[N,256]^T + bias[N].
// MODE 0: store f32. MODE 1: relu + store bf16.
// USE_TDM 0: per-lane async-LDS staging.  USE_TDM 1: TDM descriptor staging.
// ---------------------------------------------------------------------------
template <int MODE, int USE_TDM>
__global__ __launch_bounds__(256)
void snn_gemm_wmma(const __bf16* __restrict__ A,
                   const __bf16* __restrict__ BT,
                   const float* __restrict__ bias,
                   void* __restrict__ Cv,
                   int N) {
    __shared__ __align__(16) __bf16 smem[4 * SEG];   // [buf][A|B][row][k]

    const int tid  = threadIdx.x;
    const int lane = tid & 31;
    const int wave = tid >> 5;
    const int wm   = wave & 1;        // 2 waves along M (64 rows each)
    const int wn   = wave >> 1;       // 4 waves along N (32 cols each)
    const int bm   = blockIdx.y * BM;
    const int bn   = blockIdx.x * BN;
    const int half = lane >> 4;
    const int rsel = lane & 15;

    const unsigned ldsbase = (unsigned)(unsigned long long)(void*)smem;
    const unsigned long long Abase =
        (unsigned long long)(const void*)(A  + (size_t)bm * H_DIM);
    const unsigned long long Bbase =
        (unsigned long long)(const void*)(BT + (size_t)bn * H_DIM);

    // async path: 512 x 16B chunks per tile; this thread owns tid, tid+256.
    const int c0 = tid, c1 = tid + 256;
    const int r0 = c0 >> 2, s0 = (c0 & 3) << 4;
    const int r1 = c1 >> 2, s1 = (c1 & 3) << 4;

    auto issue_async = [&](int k0, int buf) {
        const unsigned aL = ldsbase + (unsigned)(buf * 2 * SEG * 2);
        const unsigned bL = aL + (unsigned)(SEG * 2);
        const unsigned kb = (unsigned)(k0 * 2);
        async_b128(aL + r0 * (LDSS * 2) + s0, (unsigned)(r0 * (H_DIM * 2)) + kb + s0, Abase);
        async_b128(aL + r1 * (LDSS * 2) + s1, (unsigned)(r1 * (H_DIM * 2)) + kb + s1, Abase);
        async_b128(bL + r0 * (LDSS * 2) + s0, (unsigned)(r0 * (H_DIM * 2)) + kb + s0, Bbase);
        async_b128(bL + r1 * (LDSS * 2) + s1, (unsigned)(r1 * (H_DIM * 2)) + kb + s1, Bbase);
    };
    auto issue_tdm = [&](int k0, int buf) {    // wave 0 only
        const unsigned aL = ldsbase + (unsigned)(buf * 2 * SEG * 2);
        const unsigned bL = aL + (unsigned)(SEG * 2);
        tdm_tile_2d(aL, Abase + (unsigned long long)(k0 * 2));
        tdm_tile_2d(bL, Bbase + (unsigned long long)(k0 * 2));
    };

    v8f acc[4][2];
#pragma unroll
    for (int mt = 0; mt < 4; ++mt)
#pragma unroll
        for (int nt = 0; nt < 2; ++nt)
            acc[mt][nt] = {};

    if (USE_TDM) { if (wave == 0) issue_tdm(0, 0); }
    else         issue_async(0, 0);

    for (int kk = 0; kk < KSTEPS; ++kk) {
        const int buf = kk & 1;
        if (USE_TDM) {
            if (wave == 0) __builtin_amdgcn_s_wait_tensorcnt(0);  // tile kk landed
            __syncthreads();                   // broadcast; prior reads done
            if (wave == 0 && kk < KSTEPS - 1)
                issue_tdm((kk + 1) * BK, buf ^ 1);
        } else {
            if (kk < KSTEPS - 1) {
                issue_async((kk + 1) * BK, buf ^ 1);
                wait_async_le4();              // tile kk done; kk+1 in flight
            } else {
                wait_async_0();
            }
            __syncthreads();
        }

        const __bf16* As = smem + buf * 2 * SEG;
        const __bf16* Bs = As + SEG;

        // A 16x32 layout: lane<16 holds K {0..7,16..23}; lane>=16 {8..15,24..31}
        v16bf afrag[4];
#pragma unroll
        for (int mt = 0; mt < 4; ++mt) {
            const __bf16* p = &As[(wm * 64 + mt * 16 + rsel) * LDSS + half * 8];
            v8bf lo = *reinterpret_cast<const v8bf*>(p);
            v8bf hi = *reinterpret_cast<const v8bf*>(p + 16);
            afrag[mt] = __builtin_shufflevector(lo, hi,
                0,1,2,3,4,5,6,7,8,9,10,11,12,13,14,15);
        }
        // B 32x16 layout: lane<16 holds K {0..7,16..23} of col n=lane; lane>=16 {8..15,24..31}
        v16bf bfrag[2];
#pragma unroll
        for (int nt = 0; nt < 2; ++nt) {
            const __bf16* p = &Bs[(wn * 32 + nt * 16 + rsel) * LDSS + half * 8];
            v8bf lo = *reinterpret_cast<const v8bf*>(p);
            v8bf hi = *reinterpret_cast<const v8bf*>(p + 16);
            bfrag[nt] = __builtin_shufflevector(lo, hi,
                0,1,2,3,4,5,6,7,8,9,10,11,12,13,14,15);
        }

#pragma unroll
        for (int mt = 0; mt < 4; ++mt)
#pragma unroll
            for (int nt = 0; nt < 2; ++nt)
                acc[mt][nt] = __builtin_amdgcn_wmma_f32_16x16x32_bf16(
                    false, afrag[mt], false, bfrag[nt],
                    (short)0, acc[mt][nt], false, false);

        if (!USE_TDM) __syncthreads();         // async path: reads before refill
    }

    // Epilogue: C/D layout -> M = tile + 8*half + r, N = tile + rsel.
#pragma unroll
    for (int mt = 0; mt < 4; ++mt) {
#pragma unroll
        for (int nt = 0; nt < 2; ++nt) {
            int n = bn + wn * 32 + nt * 16 + rsel;
            float bv = bias[n];
#pragma unroll
            for (int r = 0; r < 8; ++r) {
                int m = bm + wm * 64 + mt * 16 + half * 8 + r;
                float v = acc[mt][nt][r] + bv;
                if (MODE == 0) {
                    ((float*)Cv)[(size_t)m * N + n] = v;
                } else {
                    v = v > 0.0f ? v : 0.0f;
                    ((__bf16*)Cv)[(size_t)m * N + n] = (__bf16)v;
                }
            }
        }
    }
}

// ---------------------------------------------------------------------------
extern "C" void kernel_launch(void* const* d_in, const int* in_sizes, int n_in,
                              void* d_out, int out_size, void* d_ws, size_t ws_size,
                              hipStream_t stream) {
    (void)in_sizes; (void)n_in; (void)out_size; (void)ws_size;

    const int*   tokens = (const int*)  d_in[0];
    const float* temb   = (const float*)d_in[1];
    const float* pemb   = (const float*)d_in[2];
    const float* Wq     = (const float*)d_in[3];
    const float* bq     = (const float*)d_in[4];
    const float* beta   = (const float*)d_in[5];
    const float* Wfc    = (const float*)d_in[6];
    const float* bfc    = (const float*)d_in[7];
    const float* Wout   = (const float*)d_in[8];
    const float* bout   = (const float*)d_in[9];
    float* out = (float*)d_out;

    const size_t MH = (size_t)M_DIM * H_DIM;     // 1M elements
    const size_t HH = (size_t)H_DIM * H_DIM;     // 64K elements

    __bf16* hb    = (__bf16*)d_ws;               // [M][H] bf16      (2 MB)
    __bf16* spk   = hb + MH;                     // [M][H] bf16      (2 MB)
    float*  cur   = (float*)(spk + MH);          // [M][H] f32       (4 MB)
    __bf16* WqT   = (__bf16*)(cur + MH);         // 3x[H][H] bf16
    __bf16* WfcT  = WqT + L_DIM * HH;            // 3x[H][H] bf16
    __bf16* WoutT = WfcT + L_DIM * HH;           // [V][H] bf16      (16 MB)

    // Convert + transpose all weights to bf16 [N][K].
    for (int l = 0; l < L_DIM; ++l) {
        snn_convt_kernel<<<dim3(H_DIM / 32, H_DIM / 32), 256, 0, stream>>>(
            Wq + l * HH, WqT + l * HH, H_DIM);
        snn_convt_kernel<<<dim3(H_DIM / 32, H_DIM / 32), 256, 0, stream>>>(
            Wfc + l * HH, WfcT + l * HH, H_DIM);
    }
    snn_convt_kernel<<<dim3(V_DIM / 32, H_DIM / 32), 256, 0, stream>>>(
        Wout, WoutT, V_DIM);

    snn_embed_kernel<<<(M_DIM * H_DIM) / 256, 256, 0, stream>>>(tokens, temb, pemb, hb);

    for (int l = 0; l < L_DIM; ++l) {
        snn_gemm_wmma<0, 0><<<dim3(H_DIM / BN, M_DIM / BM), 256, 0, stream>>>(
            hb, WqT + l * HH, bq + l * H_DIM, cur, H_DIM);
        snn_scan_spike_kernel<<<(B_DIM * H_DIM) / 256, 256, 0, stream>>>(
            cur, spk, beta + l * H_DIM);
        snn_gemm_wmma<1, 0><<<dim3(H_DIM / BN, M_DIM / BM), 256, 0, stream>>>(
            spk, WfcT + l * HH, bfc + l * H_DIM, hb, H_DIM);
    }

    // Logits GEMM: TDM-staged tiles, 524MB f32 streamed out (HBM-bound floor).
    snn_gemm_wmma<0, 1><<<dim3(V_DIM / BN, M_DIM / BM), 256, 0, stream>>>(
        hb, WoutT, bout, out, V_DIM);
}
